// SymmetricContraction_62380105008111
// MI455X (gfx1250) — matrix operand order repaired and checked by reading, not compiled, not run on hardware
//
#include <hip/hip_runtime.h>
#include <hip/hip_bf16.h>

typedef __attribute__((ext_vector_type(16))) _Float16 v16h;
typedef __attribute__((ext_vector_type(8)))  _Float16 h8;
typedef __attribute__((ext_vector_type(8)))  float    v8f;

#define NNODES 2048
#define NCH    128
#define IDIM   9
#define KPOLY  219   // 9 linear + 45 sym pairs + 165 sym triples
#define KPAD   224   // padded to multiple of 32 (7 WMMA K-steps)
#define KSTR   232   // LDS stride in halfs; 464B = 116 dwords, 116%64=52, gcd(52,64)=4 -> 16 distinct banks
#define WROWS  16    // m rows padded 9 -> 16

// ---------------------------------------------------------------------------
// Kernel 1: fold U·w into symmetrized polynomial weights Wcat[c][16][224] (f16)
//   slot k<9        : linear term i=k
//   slot 9..53      : pair (i<=j), sum over both orderings
//   slot 54..218    : triple (i<=j<=l), sum over distinct permutations
// Each slot owned by exactly one thread -> deterministic (no float atomics).
// ---------------------------------------------------------------------------
__global__ void build_wcat_kernel(
    const float* __restrict__ Us1, const float* __restrict__ ws1,
    const float* __restrict__ Us2, const float* __restrict__ ws2,
    const float* __restrict__ Us3, const float* __restrict__ ws3,
    const float* __restrict__ Up1, const float* __restrict__ wp1,
    const float* __restrict__ Up2, const float* __restrict__ wp2,
    const float* __restrict__ Up3, const float* __restrict__ wp3,
    const float* __restrict__ Ud1, const float* __restrict__ wd1,
    const float* __restrict__ Ud2, const float* __restrict__ wd2,
    const float* __restrict__ Ud3, const float* __restrict__ wd3,
    _Float16* __restrict__ wcat)
{
  const int c = blockIdx.x;
  const float* U1a[3] = {Us1, Up1, Ud1}; const float* w1a[3] = {ws1, wp1, wd1};
  const float* U2a[3] = {Us2, Up2, Ud2}; const float* w2a[3] = {ws2, wp2, wd2};
  const float* U3a[3] = {Us3, Up3, Ud3}; const float* w3a[3] = {ws3, wp3, wd3};
  const int k2a[3] = {2, 3, 4};
  const int k3a[3] = {5, 8, 10};

  for (int t = threadIdx.x; t < WROWS * KPAD; t += blockDim.x) {
    const int row = t / KPAD;        // global m (0..8 valid, 9..15 zero pad)
    const int k   = t % KPAD;        // polynomial slot (219..223 zero pad)
    float val = 0.f;
    if (row < 9 && k < KPOLY) {
      const int ty = (row == 0) ? 0 : (row < 4 ? 1 : 2);      // s / p / d
      const int ml = (row == 0) ? 0 : (row < 4 ? row - 1 : row - 4);
      if (k < 9) {
        // nu=1, k-count is 1 for all types
        val = U1a[ty][ml * 9 + k] * w1a[ty][c];
      } else if (k < 54) {
        int q = k - 9, i = 0;
        while (q >= 9 - i) { q -= 9 - i; ++i; }
        const int j = i + q;                                   // i <= j
        const float* U = U2a[ty]; const float* w = w2a[ty]; const int kk = k2a[ty];
        for (int p = 0; p < kk; ++p) {
          const float wc = w[p * NCH + c];
          float us = U[((ml * 9 + i) * 9 + j) * kk + p];
          if (i != j) us += U[((ml * 9 + j) * 9 + i) * kk + p];
          val += us * wc;
        }
      } else {
        int q = k - 54, i = 0;
        for (;;) { const int cnt = (9 - i) * (10 - i) / 2; if (q < cnt) break; q -= cnt; ++i; }
        int j = i;
        while (q >= 9 - j) { q -= 9 - j; ++j; }
        const int l = j + q;                                   // i <= j <= l
        int pa[6], pb[6], pc[6]; int np;
        if (i == j && j == l) { pa[0]=i; pb[0]=i; pc[0]=i; np = 1; }
        else if (i == j) { pa[0]=i;pb[0]=i;pc[0]=l; pa[1]=i;pb[1]=l;pc[1]=i; pa[2]=l;pb[2]=i;pc[2]=i; np = 3; }
        else if (j == l) { pa[0]=i;pb[0]=j;pc[0]=j; pa[1]=j;pb[1]=i;pc[1]=j; pa[2]=j;pb[2]=j;pc[2]=i; np = 3; }
        else {
          pa[0]=i;pb[0]=j;pc[0]=l; pa[1]=i;pb[1]=l;pc[1]=j; pa[2]=j;pb[2]=i;pc[2]=l;
          pa[3]=j;pb[3]=l;pc[3]=i; pa[4]=l;pb[4]=i;pc[4]=j; pa[5]=l;pb[5]=j;pc[5]=i; np = 6;
        }
        const float* U = U3a[ty]; const float* w = w3a[ty]; const int kk = k3a[ty];
        for (int p = 0; p < kk; ++p) {
          const float wc = w[p * NCH + c];
          float us = 0.f;
          for (int o = 0; o < np; ++o)
            us += U[(((ml * 9 + pa[o]) * 9 + pb[o]) * 9 + pc[o]) * kk + p];
          val += us * wc;
        }
      }
    }
    wcat[(size_t)c * (WROWS * KPAD) + t] = (_Float16)val;
  }
}

// ---------------------------------------------------------------------------
// Kernel 2: per-channel GEMM  out[16m x 16n] = Wcat[c] (16x224) @ X (224x16)
// 4 waves / block, each wave owns a 16-node tile; monomial vector X built in
// LDS by the wave's own lanes, then 7 x v_wmma_f32_16x16x32_f16.
// ---------------------------------------------------------------------------
__global__ __launch_bounds__(128) void poly_wmma_kernel(
    const float* __restrict__ A, const _Float16* __restrict__ wcat,
    float* __restrict__ out)
{
  __shared__ __attribute__((aligned(16))) _Float16 sW[WROWS * KSTR];     //  7.4 KB
  __shared__ __attribute__((aligned(16))) _Float16 sX[4 * 16 * KSTR];    // 29.7 KB

  const int tid  = threadIdx.x;
  const int c    = blockIdx.x;
  const int nb   = blockIdx.y;                 // 64-node superblock
  const int lane = tid & 31;
  const int wave = tid >> 5;                   // 0..3
  const int ncol = lane & 15;                  // node column / matrix row M
  const int half = lane >> 4;                  // fragment half select

  // Stage Wcat[c] into LDS with skewed stride.
  for (int t = tid; t < WROWS * KPAD; t += 128) {
    const int r = t / KPAD, k = t % KPAD;
    sW[r * KSTR + k] = wcat[(size_t)c * (WROWS * KPAD) + t];
  }

  // Build the symmetric-monomial vector for this lane's node.
  const int node = nb * 64 + wave * 16 + ncol;
  const float* Ap = A + ((size_t)node * NCH + c) * IDIM;
  float a[9];
  #pragma unroll
  for (int i = 0; i < 9; ++i) a[i] = Ap[i];

  _Float16* xc = sX + (wave * 16 + ncol) * KSTR;
  if (half == 0) {
    #pragma unroll
    for (int i = 0; i < 9; ++i) xc[i] = (_Float16)a[i];
    #pragma unroll
    for (int p = KPOLY; p < KPAD; ++p) xc[p] = (_Float16)0.f;  // zero pad
  } else {
    int q = 0;
    for (int i = 0; i < 9; ++i)
      for (int j = i; j < 9; ++j) { xc[9 + q] = (_Float16)(a[i] * a[j]); ++q; }
  }
  {
    int q = 0;
    for (int i = 0; i < 9; ++i)
      for (int j = i; j < 9; ++j) {
        const float pij = a[i] * a[j];
        for (int l = j; l < 9; ++l) {
          if ((q & 1) == half) xc[54 + q] = (_Float16)(pij * a[l]);
          ++q;
        }
      }
  }
  __syncthreads();

  // WMMA: A-frag lane(M=ncol): K chunks [k0+half*8, +8) and [k0+16+half*8, +8)
  //       B-frag lane(N=ncol): K chunks [k0+half*8, +8) and [k0+16+half*8, +8)
  const _Float16* wr = sW + ncol * KSTR;
  v8f acc = {};
  #pragma unroll
  for (int k0 = 0; k0 < KPAD; k0 += 32) {
    h8 alo = *(const h8*)(wr + k0 + half * 8);
    h8 ahi = *(const h8*)(wr + k0 + 16 + half * 8);
    h8 blo = *(const h8*)(xc + k0 + half * 8);
    h8 bhi = *(const h8*)(xc + k0 + 16 + half * 8);
    v16h af = __builtin_shufflevector(alo, ahi, 0,1,2,3,4,5,6,7,8,9,10,11,12,13,14,15);
    v16h bf = __builtin_shufflevector(blo, bhi, 0,1,2,3,4,5,6,7,8,9,10,11,12,13,14,15);
    acc = __builtin_amdgcn_wmma_f32_16x16x32_f16(false, af, false, bf,
                                                 (short)0, acc, false, false);
  }

  // D layout: VGPR r -> lanes 0-15: (M=r, N=lane); lanes 16-31: (M=8+r, N=lane-16)
  const size_t ob = ((size_t)node * NCH + c) * IDIM;
  if (half == 0) {
    #pragma unroll
    for (int r = 0; r < 8; ++r) out[ob + r] = acc[r];          // m = 0..7
  } else {
    out[ob + 8] = acc[0];                                      // m = 8
  }
}

extern "C" void kernel_launch(void* const* d_in, const int* in_sizes, int n_in,
                              void* d_out, int out_size, void* d_ws, size_t ws_size,
                              hipStream_t stream) {
  const float* A   = (const float*)d_in[0];
  const float* Us1 = (const float*)d_in[1];  const float* ws1 = (const float*)d_in[2];
  const float* Us2 = (const float*)d_in[3];  const float* ws2 = (const float*)d_in[4];
  const float* Us3 = (const float*)d_in[5];  const float* ws3 = (const float*)d_in[6];
  const float* Up1 = (const float*)d_in[7];  const float* wp1 = (const float*)d_in[8];
  const float* Up2 = (const float*)d_in[9];  const float* wp2 = (const float*)d_in[10];
  const float* Up3 = (const float*)d_in[11]; const float* wp3 = (const float*)d_in[12];
  const float* Ud1 = (const float*)d_in[13]; const float* wd1 = (const float*)d_in[14];
  const float* Ud2 = (const float*)d_in[15]; const float* wd2 = (const float*)d_in[16];
  const float* Ud3 = (const float*)d_in[17]; const float* wd3 = (const float*)d_in[18];

  _Float16* wcat = (_Float16*)d_ws;   // 128 * 16 * 224 * 2B = 917504 bytes

  build_wcat_kernel<<<dim3(NCH), dim3(256), 0, stream>>>(
      Us1, ws1, Us2, ws2, Us3, ws3,
      Up1, wp1, Up2, wp2, Up3, wp3,
      Ud1, wd1, Ud2, wd2, Ud3, wd3, wcat);

  poly_wmma_kernel<<<dim3(NCH, NNODES / 64), dim3(128), 0, stream>>>(
      A, wcat, (float*)d_out);
}
